// HierarchicalSoftmaxTree_23897198035513
// MI455X (gfx1250) — matrix operand order repaired and checked by reading, not compiled, not run on hardware
//
#include <hip/hip_runtime.h>
#include <hip/hip_bf16.h>
#include <math.h>

// Problem constants (match reference)
#define V_  50257
#define N_  50256
#define D_  1024
#define L_  20
#define B_  64
#define K_  32

typedef __attribute__((ext_vector_type(16))) _Float16 v16h;
typedef __attribute__((ext_vector_type(8)))  float    v8f;

// One 64-thread block (2 wave32) per (b,k) target.
// Wave w computes path positions [16w, 16w+16) as the N dimension of
// v_wmma_f32_16x16x32_f16 tiles. Every A row carries the same broadcast
// x_b chunk (rows are independent in WMMA, so no zero-masking is needed);
// C[0][n] (VGPR0 of lanes 0..15) accumulates dot(x_b, W[path[n]]).
__global__ __launch_bounds__(64) void hsm_wmma_kernel(
    const float* __restrict__ x,              // [B, D]
    const int*   __restrict__ ids,            // [B, K]
    const float* __restrict__ W,              // [N, D]
    const float* __restrict__ bias,           // [N]
    const int*   __restrict__ paths,          // [V, L]
    const unsigned char* __restrict__ pmask,  // [V, L] (bool)
    float* __restrict__ out)                  // [B, K]
{
    __shared__ _Float16      sx[D_];      // x_b as f16 (2 KB)
    __shared__ int           srow[32];    // gathered node rows (padded)
    __shared__ float         sbias[32];
    __shared__ unsigned char smask[32];
    __shared__ float         spart[2];

    const int tid  = threadIdx.x;
    const int lane = tid & 31;
    const int wave = tid >> 5;
    const int brow = blockIdx.x >> 5;     // K == 32
    const int id   = ids[blockIdx.x];

    // ---- stage path rows / masks / biases (32 tile columns, 20 valid) ----
    if (tid < 32) {
        const int l    = tid;
        const int srcl = (l < L_) ? l : (16 + (l & 3)); // pad -> alias rows 16..19
        const int row  = paths[(size_t)id * L_ + srcl];
        srow[tid]  = row;
        smask[tid] = (l < L_) ? pmask[(size_t)id * L_ + l] : (unsigned char)0;
        sbias[tid] = bias[row];
        // Warm L2 with the head of each gathered row (global_prefetch_b8).
        __builtin_prefetch(W + (size_t)row * D_, 0, 1);
    }

    // ---- stage x_b into LDS as f16 (float4 granularity) ----
    {
        const float4* xv = (const float4*)(x + (size_t)brow * D_);
        for (int j = tid; j < D_ / 4; j += 64) {
            float4 v = xv[j];
            sx[4 * j + 0] = (_Float16)v.x;
            sx[4 * j + 1] = (_Float16)v.y;
            sx[4 * j + 2] = (_Float16)v.z;
            sx[4 * j + 3] = (_Float16)v.w;
        }
    }
    __syncthreads();

    const int col = lane & 15;            // tile column (path position within wave)
    const int hi  = (lane >> 4) & 1;      // K-half select (0: K=0..15, 1: K=16..31)
    const int row = srow[wave * 16 + col];
    const float* __restrict__ wp = W + (size_t)row * D_ + (hi << 4);

    v8f c = {};
    #pragma unroll 4
    for (int d0 = 0; d0 < D_; d0 += 32) {
        // B fragment: 16 consecutive f32 of this column's W row -> f16.
        const float4* w4 = (const float4*)(wp + d0);
        float4 f0 = w4[0], f1 = w4[1], f2 = w4[2], f3 = w4[3];
        v16h bfr;
        bfr[0]  = (_Float16)f0.x; bfr[1]  = (_Float16)f0.y;
        bfr[2]  = (_Float16)f0.z; bfr[3]  = (_Float16)f0.w;
        bfr[4]  = (_Float16)f1.x; bfr[5]  = (_Float16)f1.y;
        bfr[6]  = (_Float16)f1.z; bfr[7]  = (_Float16)f1.w;
        bfr[8]  = (_Float16)f2.x; bfr[9]  = (_Float16)f2.y;
        bfr[10] = (_Float16)f2.z; bfr[11] = (_Float16)f2.w;
        bfr[12] = (_Float16)f3.x; bfr[13] = (_Float16)f3.y;
        bfr[14] = (_Float16)f3.z; bfr[15] = (_Float16)f3.w;

        // A fragment (16x32 f16 layout): every row = broadcast x_b chunk.
        // Lane m holds K 0..15; lane m+16 holds K 16..31.
        // Rows are independent in WMMA, so duplicated rows are harmless;
        // we only consume C row 0.
        const int base = d0 + (hi ? 16 : 0);
        v16h afr;
        #pragma unroll
        for (int i = 0; i < 8; ++i) {     // broadcast LDS reads (same addr within each 16-lane half)
            afr[i]     = sx[base + i];
            afr[8 + i] = sx[base + 8 + i];
        }

        c = __builtin_amdgcn_wmma_f32_16x16x32_f16(
                /*neg_a=*/false, afr, /*neg_b=*/false, bfr,
                /*c_mod=*/(short)0, c, /*reuse_a=*/false, /*reuse_b=*/false);
    }

    // ---- epilogue: bias + masked sigmoid, product over path positions ----
    float p = 1.0f;
    if (lane < 16) {
        const int t = wave * 16 + lane;
        if (t < L_ && smask[t]) {
            const float logit = c[0] + sbias[t];   // C row 0, N = lane
            p = __fdividef(1.0f, 1.0f + __expf(-logit));
        }
    }
    #pragma unroll
    for (int off = 16; off > 0; off >>= 1)
        p *= __shfl_xor(p, off, 32);
    if (lane == 0) spart[wave] = p;
    __syncthreads();
    if (tid == 0) out[blockIdx.x] = spart[0] * spart[1];
}

extern "C" void kernel_launch(void* const* d_in, const int* in_sizes, int n_in,
                              void* d_out, int out_size, void* d_ws, size_t ws_size,
                              hipStream_t stream) {
    (void)in_sizes; (void)n_in; (void)out_size; (void)d_ws; (void)ws_size;
    const float*         x     = (const float*)d_in[0];
    const int*           ids   = (const int*)d_in[1];
    const float*         W     = (const float*)d_in[2];
    const float*         bias  = (const float*)d_in[3];
    const int*           paths = (const int*)d_in[4];
    const unsigned char* pmask = (const unsigned char*)d_in[5];
    float*               out   = (float*)d_out;

    dim3 grid(B_ * K_);   // 2048 blocks, one per (b,k)
    dim3 block(64);       // 2 wave32 per block
    hipLaunchKernelGGL(hsm_wmma_kernel, grid, block, 0, stream,
                       x, ids, W, bias, paths, pmask, out);
}
